// MLP_BFR_S_6107443495226
// MI455X (gfx1250) — matrix operand order repaired and checked by reading, not compiled, not run on hardware
//
#include <hip/hip_runtime.h>
#include <math.h>

// ---------- types ----------
typedef __bf16 v16bf __attribute__((ext_vector_type(16)));
typedef float  v8f   __attribute__((ext_vector_type(8)));
typedef unsigned int u32x4 __attribute__((ext_vector_type(4)));
typedef unsigned short ushort_t;

union FragBF { v16bf v; u32x4 q[2]; };   // 32 bytes: one WMMA A/B operand per lane

// ---------- problem constants ----------
#define C_B    256      // batch
#define C_N    4096     // nodes
#define C_E    131072   // edges
#define C_K1   16384    // 4*N  (GEMM1 K)
#define C_NH   2048     // hidden
#define C_OUT  256
#define C_ALPHA 1e-4f
#define C_BETA  1e-4f

// ---------- bf16 split helpers ----------
__device__ __forceinline__ ushort_t f2bf(float f) {
  unsigned u = __float_as_uint(f);
  u += 0x7FFFu + ((u >> 16) & 1u);        // round-to-nearest-even
  return (ushort_t)(u >> 16);
}
__device__ __forceinline__ float bf2f(ushort_t h) {
  return __uint_as_float(((unsigned)h) << 16);
}

// Workgroup barrier with LDS-only visibility: waits DScnt but leaves global
// loads in flight (unlike __syncthreads(), whose fence drains loadcnt to 0).
__device__ __forceinline__ void wg_barrier_lds() {
  asm volatile("s_wait_dscnt 0x0\n\t"
               "s_barrier_signal -1\n\t"
               "s_barrier_wait -1" ::: "memory");
}

// ============================================================
// 0) fp32 tile transpose: in[R][C] -> out[C][R]   (R,C multiples of 64)
// ============================================================
__global__ __launch_bounds__(256) void transpose_f32(const float* __restrict__ in,
                                                     float* __restrict__ out,
                                                     int R, int C) {
  __shared__ float tile[64][65];
  int r0 = blockIdx.x * 64, c0 = blockIdx.y * 64;
  for (int i = threadIdx.x; i < 64 * 64; i += 256) {
    int r = i >> 6, c = i & 63;
    tile[r][c] = in[(size_t)(r0 + r) * C + (c0 + c)];
  }
  __syncthreads();
  for (int i = threadIdx.x; i < 64 * 64; i += 256) {
    int c = i >> 6, r = i & 63;
    out[(size_t)(c0 + c) * R + (r0 + r)] = tile[r][c];
  }
}

// ============================================================
// 1) transpose + split fp32 -> (hi,lo) bf16 : W[K][Nn] -> T[Nn][K]
// ============================================================
__global__ __launch_bounds__(256) void split_transpose(const float* __restrict__ W,
                                                       ushort_t* __restrict__ hiT,
                                                       ushort_t* __restrict__ loT,
                                                       int K, int Nn) {
  __shared__ float tile[64][65];
  int k0 = blockIdx.x * 64, j0 = blockIdx.y * 64;
  for (int i = threadIdx.x; i < 64 * 64; i += 256) {
    int r = i >> 6, c = i & 63;
    tile[r][c] = W[(size_t)(k0 + r) * Nn + (j0 + c)];
  }
  __syncthreads();
  for (int i = threadIdx.x; i < 64 * 64; i += 256) {
    int j = i >> 6, kk = i & 63;
    float v = tile[kk][j];
    ushort_t hi = f2bf(v);
    ushort_t lo = f2bf(v - bf2f(hi));
    size_t idx = (size_t)(j0 + j) * K + (k0 + kk);
    hiT[idx] = hi;
    loT[idx] = lo;
  }
}

// ============================================================
// 2) CSR build: degree count, scan, fill
// ============================================================
__global__ void count_deg(const int* __restrict__ edst, int* __restrict__ degi) {
  int e = blockIdx.x * blockDim.x + threadIdx.x;
  if (e < C_E) atomicAdd(&degi[edst[e]], 1);
}

__global__ __launch_bounds__(1024) void scan_deg(const int* __restrict__ degi,
                                                 int* __restrict__ rowoff,
                                                 int* __restrict__ cursor,
                                                 float* __restrict__ rdeg) {
  __shared__ int ss[1024];
  int t = threadIdx.x;
  int base = t * 4;
  int v0 = degi[base + 0], v1 = degi[base + 1], v2 = degi[base + 2], v3 = degi[base + 3];
  int s = v0 + v1 + v2 + v3;
  ss[t] = s;
  __syncthreads();
  for (int off = 1; off < 1024; off <<= 1) {
    int add = (t >= off) ? ss[t - off] : 0;
    __syncthreads();
    ss[t] += add;
    __syncthreads();
  }
  int excl = ss[t] - s;
  int o0 = excl, o1 = o0 + v0, o2 = o1 + v1, o3 = o2 + v2;
  rowoff[base + 0] = o0; rowoff[base + 1] = o1; rowoff[base + 2] = o2; rowoff[base + 3] = o3;
  cursor[base + 0] = o0; cursor[base + 1] = o1; cursor[base + 2] = o2; cursor[base + 3] = o3;
  rdeg[base + 0] = 1.0f / (float)(v0 > 1 ? v0 : 1);
  rdeg[base + 1] = 1.0f / (float)(v1 > 1 ? v1 : 1);
  rdeg[base + 2] = 1.0f / (float)(v2 > 1 ? v2 : 1);
  rdeg[base + 3] = 1.0f / (float)(v3 > 1 ? v3 : 1);
  if (t == 1023) rowoff[C_N] = ss[1023];
}

__global__ void fill_csr(const int* __restrict__ esrc, const int* __restrict__ edst,
                         int* __restrict__ cursor, int* __restrict__ nbr) {
  int e = blockIdx.x * blockDim.x + threadIdx.x;
  if (e < C_E) {
    int d = edst[e];
    int pos = atomicAdd(&cursor[d], 1);
    nbr[pos] = esrc[e];
  }
}

// ============================================================
// 3) message-passing pass 1: h = elu((mean_agg(x)+a*x) @ Wg1 + bg1)
//    xT: [N][B] f32 ; hh: [N][4][B] f32
// ============================================================
__global__ __launch_bounds__(256) void mp_pass1(const float* __restrict__ xT,
                                                const int* __restrict__ rowoff,
                                                const int* __restrict__ nbr,
                                                const float* __restrict__ rdeg,
                                                const float* __restrict__ Wg1,
                                                const float* __restrict__ bg1,
                                                float* __restrict__ hh) {
  int n = blockIdx.x;
  int b = threadIdx.x;
  int beg = rowoff[n], end = rowoff[n + 1];
  float s = 0.f;
  for (int i = beg; i < end; ++i) s += xT[(size_t)nbr[i] * C_B + b];
  float t = s * rdeg[n] + C_ALPHA * xT[(size_t)n * C_B + b];
#pragma unroll
  for (int c = 0; c < 4; ++c) {
    float hv = fmaf(t, Wg1[c], bg1[c]);
    hv = hv > 0.f ? hv : expm1f(hv);        // elu
    hh[((size_t)n * 4 + c) * C_B + b] = hv;
  }
}

// ============================================================
// 4) pass 2: g = (mean_agg(h)+b*h) @ Wg2 + bg2 ; write A as split bf16,
//    row-major [B][4N] (K-contiguous -> WMMA-A friendly)
// ============================================================
__global__ __launch_bounds__(256) void mp_pass2(const float* __restrict__ hh,
                                                const int* __restrict__ rowoff,
                                                const int* __restrict__ nbr,
                                                const float* __restrict__ rdeg,
                                                const float* __restrict__ Wg2,
                                                const float* __restrict__ bg2,
                                                ushort_t* __restrict__ Ahi,
                                                ushort_t* __restrict__ Alo) {
  int n = blockIdx.x;
  int b = threadIdx.x;
  int beg = rowoff[n], end = rowoff[n + 1];
  float a0 = 0.f, a1 = 0.f, a2 = 0.f, a3 = 0.f;
  for (int i = beg; i < end; ++i) {
    size_t base = (size_t)nbr[i] * (4 * C_B) + b;
    a0 += hh[base + 0 * C_B];
    a1 += hh[base + 1 * C_B];
    a2 += hh[base + 2 * C_B];
    a3 += hh[base + 3 * C_B];
  }
  float rd = rdeg[n];
  size_t nb = (size_t)n * (4 * C_B) + b;
  float t0 = a0 * rd + C_BETA * hh[nb + 0 * C_B];
  float t1 = a1 * rd + C_BETA * hh[nb + 1 * C_B];
  float t2 = a2 * rd + C_BETA * hh[nb + 2 * C_B];
  float t3 = a3 * rd + C_BETA * hh[nb + 3 * C_B];
#pragma unroll
  for (int c2 = 0; c2 < 4; ++c2) {
    float g = bg2[c2] + t0 * Wg2[0 * 4 + c2] + t1 * Wg2[1 * 4 + c2]
                      + t2 * Wg2[2 * 4 + c2] + t3 * Wg2[3 * 4 + c2];
    ushort_t hi = f2bf(g);
    ushort_t lo = f2bf(g - bf2f(hi));
    size_t idx = (size_t)b * C_K1 + (size_t)n * 4 + c2;   // A[b][4n+c2]
    Ahi[idx] = hi;
    Alo[idx] = lo;
  }
}

// ============================================================
// 5) split-bf16 WMMA GEMM (K-split partials):
//    part[z][M][Nn] += A[M][kz..kz+Kc] * BT[Nn][kz..kz+Kc]^T
//    3 WMMAs per tile (hi*hi + hi*lo + lo*hi) ~ fp32 accuracy.
//    Block tile 64(M) x 128(N), BK=32, 256 threads (8 waves, 2x2 subtiles/wave).
//    2-stage software pipeline (sched_barrier-pinned) + double-buffered LDS +
//    dscnt-only barriers so stage loads stay in flight across the barrier.
// ============================================================
#define BM 64
#define BN 128
#define BK 32
#define BKP 48   // padded LDS row stride (ushort units): 96B, 16B-aligned rows

__device__ __forceinline__ v8f wmma_bf16(const FragBF& a, const FragBF& b, v8f c) {
  return __builtin_amdgcn_wmma_f32_16x16x32_bf16(false, a.v, false, b.v,
                                                 (short)0, c, false, false);
}

__global__ __launch_bounds__(256) void gemm_split(const ushort_t* __restrict__ Ahi,
                                                  const ushort_t* __restrict__ Alo,
                                                  const ushort_t* __restrict__ BThi,
                                                  const ushort_t* __restrict__ BTlo,
                                                  int M, int K, int Nn,
                                                  float* __restrict__ part) {
  __shared__ ushort_t AsHi[2][BM][BKP], AsLo[2][BM][BKP];
  __shared__ ushort_t BsHi[2][BN][BKP], BsLo[2][BN][BKP];

  int tid = threadIdx.x;
  int lane = tid & 31, wid = tid >> 5;
  int wm = wid & 1, wn = wid >> 1;            // 2 waves along M, 4 along N
  int m0 = blockIdx.x * BM, n0 = blockIdx.y * BN;

  // K-range of this split
  int Kc = K / gridDim.z;
  int kbeg = blockIdx.z * Kc;
  int kend = kbeg + Kc;

  v8f zero = {0.f, 0.f, 0.f, 0.f, 0.f, 0.f, 0.f, 0.f};
  v8f acc[2][2];
  acc[0][0] = zero; acc[0][1] = zero; acc[1][0] = zero; acc[1][1] = zero;

  int arow = tid >> 2;             // 0..63
  int achunk = (tid & 3) * 8;      // 0,8,16,24  (ushort units -> 16B chunks)
  int h = lane >> 4, mr = lane & 15;

  // per-thread base pointers for the cooperative tile loads
  const ushort_t* gA0 = &Ahi [(size_t)(m0 + arow) * K + achunk];
  const ushort_t* gA1 = &Alo [(size_t)(m0 + arow) * K + achunk];
  const ushort_t* gB0 = &BThi[(size_t)(n0 + arow) * K + achunk];
  const ushort_t* gB1 = &BThi[(size_t)(n0 + arow + 64) * K + achunk];
  const ushort_t* gB2 = &BTlo[(size_t)(n0 + arow) * K + achunk];
  const ushort_t* gB3 = &BTlo[(size_t)(n0 + arow + 64) * K + achunk];

  // issue all 6 tile loads for K-offset kk into a register stage (no waits here)
  auto gload = [&](u32x4* st, int kk) {
    st[0] = *(const u32x4*)(gA0 + kk);
    st[1] = *(const u32x4*)(gA1 + kk);
    st[2] = *(const u32x4*)(gB0 + kk);
    st[3] = *(const u32x4*)(gB1 + kk);
    st[4] = *(const u32x4*)(gB2 + kk);
    st[5] = *(const u32x4*)(gB3 + kk);
    if (kk + BK < kend) {   // pull W stream 2 tiles ahead into L2 (global_prefetch_b8)
      __builtin_prefetch(gB0 + kk + BK, 0, 1);
      __builtin_prefetch(gB2 + kk + BK, 0, 1);
    }
  };
  // drain a register stage into LDS buffer p (threshold loadcnt wait lands here)
  auto lstore = [&](const u32x4* st, int p) {
    *(u32x4*)&AsHi[p][arow][achunk]      = st[0];
    *(u32x4*)&AsLo[p][arow][achunk]      = st[1];
    *(u32x4*)&BsHi[p][arow][achunk]      = st[2];
    *(u32x4*)&BsHi[p][arow + 64][achunk] = st[3];
    *(u32x4*)&BsLo[p][arow][achunk]      = st[4];
    *(u32x4*)&BsLo[p][arow + 64][achunk] = st[5];
  };
  // fragment loads (ISA 16-bit A/B VGPR layouts) + 12 WMMAs from buffer p
  auto compute = [&](int p) {
    FragBF aHi[2], aLo[2], bHi[2], bLo[2];
#pragma unroll
    for (int sm = 0; sm < 2; ++sm) {
      int row = wm * 32 + sm * 16 + mr;
      aHi[sm].q[0] = *(const u32x4*)&AsHi[p][row][h * 8];       // k = 8h..8h+7
      aHi[sm].q[1] = *(const u32x4*)&AsHi[p][row][16 + h * 8];  // k = 16+8h..
      aLo[sm].q[0] = *(const u32x4*)&AsLo[p][row][h * 8];
      aLo[sm].q[1] = *(const u32x4*)&AsLo[p][row][16 + h * 8];
    }
#pragma unroll
    for (int sn = 0; sn < 2; ++sn) {
      int row = wn * 32 + sn * 16 + mr;
      bHi[sn].q[0] = *(const u32x4*)&BsHi[p][row][h * 8];       // k = 8h..8h+7
      bHi[sn].q[1] = *(const u32x4*)&BsHi[p][row][16 + h * 8];  // k = 16+8h..
      bLo[sn].q[0] = *(const u32x4*)&BsLo[p][row][h * 8];
      bLo[sn].q[1] = *(const u32x4*)&BsLo[p][row][16 + h * 8];
    }
#pragma unroll
    for (int sm = 0; sm < 2; ++sm)
#pragma unroll
      for (int sn = 0; sn < 2; ++sn) {
        acc[sm][sn] = wmma_bf16(aHi[sm], bHi[sn], acc[sm][sn]);
        acc[sm][sn] = wmma_bf16(aHi[sm], bLo[sn], acc[sm][sn]);
        acc[sm][sn] = wmma_bf16(aLo[sm], bHi[sn], acc[sm][sn]);
      }
  };

  // ---- 2-stage pipeline, manually 2x-unrolled (Kc/BK is even: 256 or 16) ----
  u32x4 st0[6], st1[6];
  gload(st0, kbeg);
  int k0 = kbeg;
  while (true) {
    if (k0 + BK < kend) gload(st1, k0 + BK);   // next tile in flight (stage 1)
    lstore(st0, 0);                             // waits on stage-0 loads only
    wg_barrier_lds();                           // dscnt-only barrier
    __builtin_amdgcn_sched_barrier(0);          // pin: stage-1 loads stay above
    compute(0);
    k0 += BK;
    if (k0 >= kend) break;

    if (k0 + BK < kend) gload(st0, k0 + BK);   // next tile in flight (stage 0)
    lstore(st1, 1);
    wg_barrier_lds();                           // dscnt-only barrier
    __builtin_amdgcn_sched_barrier(0);          // pin: stage-0 loads stay above
    compute(1);
    k0 += BK;
    if (k0 >= kend) break;
  }

  // ---- write fp32 partial sums (no bias here; reduce kernel fuses it) ----
  float* pz = part + (size_t)blockIdx.z * M * Nn;
#pragma unroll
  for (int sm = 0; sm < 2; ++sm)
#pragma unroll
    for (int sn = 0; sn < 2; ++sn) {
      int n = n0 + wn * 32 + sn * 16 + (lane & 15);
#pragma unroll
      for (int r = 0; r < 8; ++r) {
        int m = m0 + wm * 32 + sm * 16 + r + 8 * (lane >> 4);
        pz[(size_t)m * Nn + n] = acc[sm][sn][r];
      }
    }
}

// ============================================================
// 6) K-split reduction + bias + optional elu; outputs fp32 or split-bf16
// ============================================================
__global__ __launch_bounds__(256) void reduce_fuse(const float* __restrict__ part,
                                                   int splits, int total, int Nn,
                                                   const float* __restrict__ bias,
                                                   int mode,
                                                   float* __restrict__ outF,
                                                   ushort_t* __restrict__ outHi,
                                                   ushort_t* __restrict__ outLo) {
  int i = blockIdx.x * 256 + threadIdx.x;
  if (i >= total) return;
  float v = 0.f;
  for (int s = 0; s < splits; ++s) v += part[(size_t)s * total + i];
  v += bias[i % Nn];
  if (mode == 1) {
    v = v > 0.f ? v : expm1f(v);               // elu
    ushort_t hi = f2bf(v);
    ushort_t lo = f2bf(v - bf2f(hi));
    outHi[i] = hi;
    outLo[i] = lo;
  } else {
    outF[i] = v;
  }
}

// ============================================================
// host-side launcher
// ============================================================
extern "C" void kernel_launch(void* const* d_in, const int* in_sizes, int n_in,
                              void* d_out, int out_size, void* d_ws, size_t ws_size,
                              hipStream_t stream) {
  (void)in_sizes; (void)n_in; (void)out_size; (void)ws_size;
  const float* x   = (const float*)d_in[0];
  const int* esrc  = (const int*)d_in[1];
  const int* edst  = (const int*)d_in[2];
  const float* Wg1 = (const float*)d_in[3];
  const float* bg1 = (const float*)d_in[4];
  const float* Wg2 = (const float*)d_in[5];
  const float* bg2 = (const float*)d_in[6];
  const float* W1  = (const float*)d_in[7];
  const float* b1  = (const float*)d_in[8];
  const float* W2  = (const float*)d_in[9];
  const float* b2  = (const float*)d_in[10];
  float* out = (float*)d_out;

  char* ws = (char*)d_ws;
  size_t off = 0;
  auto alloc = [&](size_t bytes) -> char* {
    char* p = ws + off;
    off = (off + bytes + 255) & ~(size_t)255;
    return p;
  };
  float*    xT     = (float*)alloc((size_t)C_N * C_B * 4);          //  4 MB
  float*    hh     = (float*)alloc((size_t)C_N * 4 * C_B * 4);      // 16 MB
  int*      degi   = (int*)alloc((size_t)C_N * 4);
  int*      rowoff = (int*)alloc((size_t)(C_N + 1) * 4);
  int*      cursor = (int*)alloc((size_t)C_N * 4);
  float*    rdeg   = (float*)alloc((size_t)C_N * 4);
  int*      nbr    = (int*)alloc((size_t)C_E * 4);                  // 0.5 MB
  ushort_t* Ahi    = (ushort_t*)alloc((size_t)C_B * C_K1 * 2);      //  8 MB
  ushort_t* Alo    = (ushort_t*)alloc((size_t)C_B * C_K1 * 2);      //  8 MB
  ushort_t* W1hiT  = (ushort_t*)alloc((size_t)C_NH * C_K1 * 2);     // 64 MB
  ushort_t* W1loT  = (ushort_t*)alloc((size_t)C_NH * C_K1 * 2);     // 64 MB
  ushort_t* W2hiT  = (ushort_t*)alloc((size_t)C_OUT * C_NH * 2);    //  1 MB
  ushort_t* W2loT  = (ushort_t*)alloc((size_t)C_OUT * C_NH * 2);    //  1 MB
  ushort_t* Y1hi   = (ushort_t*)alloc((size_t)C_B * C_NH * 2);      //  1 MB
  ushort_t* Y1lo   = (ushort_t*)alloc((size_t)C_B * C_NH * 2);      //  1 MB
  float*    part1  = (float*)alloc((size_t)2 * C_B * C_NH * 4);     //  4 MB
  float*    part2  = (float*)alloc((size_t)4 * C_B * C_OUT * 4);    //  1 MB

  // --- graph structure ---
  hipMemsetAsync(degi, 0, (size_t)C_N * 4, stream);
  transpose_f32<<<dim3(C_B / 64, C_N / 64), 256, 0, stream>>>(x, xT, C_B, C_N);
  count_deg<<<C_E / 256, 256, 0, stream>>>(edst, degi);
  scan_deg<<<1, 1024, 0, stream>>>(degi, rowoff, cursor, rdeg);
  fill_csr<<<C_E / 256, 256, 0, stream>>>(esrc, edst, cursor, nbr);

  // --- weight conversion (independent; streams W1 once) ---
  split_transpose<<<dim3(C_K1 / 64, C_NH / 64), 256, 0, stream>>>(W1, W1hiT, W1loT, C_K1, C_NH);
  split_transpose<<<dim3(C_NH / 64, C_OUT / 64), 256, 0, stream>>>(W2, W2hiT, W2loT, C_NH, C_OUT);

  // --- message passing ---
  mp_pass1<<<C_N, 256, 0, stream>>>(xT, rowoff, nbr, rdeg, Wg1, bg1, hh);
  mp_pass2<<<C_N, 256, 0, stream>>>(hh, rowoff, nbr, rdeg, Wg2, bg2, Ahi, Alo);

  // --- MLP head: GEMM1 (K-split x2 -> 128 WGs), reduce+elu, GEMM2, reduce ---
  gemm_split<<<dim3(C_B / BM, C_NH / BN, 2), 256, 0, stream>>>(
      Ahi, Alo, W1hiT, W1loT, C_B, C_K1, C_NH, part1);
  reduce_fuse<<<(C_B * C_NH) / 256, 256, 0, stream>>>(
      part1, 2, C_B * C_NH, C_NH, b1, /*mode=*/1, nullptr, Y1hi, Y1lo);
  gemm_split<<<dim3(C_B / BM, C_OUT / BN, 4), 256, 0, stream>>>(
      Y1hi, Y1lo, W2hiT, W2loT, C_B, C_NH, C_OUT, part2);
  reduce_fuse<<<(C_B * C_OUT) / 256, 256, 0, stream>>>(
      part2, 4, C_B * C_OUT, C_OUT, b2, /*mode=*/0, out, nullptr, nullptr);
}
